// PAM_Module_47562467836091
// MI455X (gfx1250) — matrix operand order repaired and checked by reading, not compiled, not run on hardware
//
#include <hip/hip_runtime.h>
#include <stdint.h>

// PAM (position attention) for B=8, C=128, H=W=64 (N=4096), fp32 in/out.
// fp32 projection -> f16 staging (Qt/Kt [N][C], V [C][N]) -> flash-attention
// with v_wmma_f32_16x16x32_f16, double-buffered K/V tiles streamed with
// GLOBAL_LOAD_ASYNC_TO_LDS_B128 (ASYNCcnt), fused softmax + residual.

constexpr int kB = 8;
constexpr int kC = 128;
constexpr int kN = 4096;
constexpr int kTM = 64;   // m-tile width
constexpr int kBN = 128;  // n rows per block = 8 waves * 16

typedef __attribute__((ext_vector_type(16))) _Float16 v16h;
typedef __attribute__((ext_vector_type(8)))  _Float16 v8h;
typedef __attribute__((ext_vector_type(8)))  float    v8f;

// A-matrix 16x32 f16 fragment (ISA 7.12.2): lane l holds row (l&15);
// element e -> K = (e<8 ? 8*h+e : 16+8*h+(e-8)), h = l>>4.
static __device__ __forceinline__ v16h frag_a(const _Float16* row, int kbase, int h) {
  v8h lo = *(const v8h*)(row + kbase + 8 * h);
  v8h hi = *(const v8h*)(row + kbase + 16 + 8 * h);
  return __builtin_shufflevector(lo, hi, 0,1,2,3,4,5,6,7,8,9,10,11,12,13,14,15);
}

// B-matrix 32x16 f16 fragment: lane l holds col (l&15); element e -> K = (e<8 ? 8*h+e : 16+8*h+(e-8)).
// With B stored transposed ([n][k] row-major) this is two 16B loads.
static __device__ __forceinline__ v16h frag_b(const _Float16* row, int kbase, int h) {
  v8h lo = *(const v8h*)(row + kbase + 8 * h);
  v8h hi = *(const v8h*)(row + kbase + 16 + 8 * h);
  return __builtin_shufflevector(lo, hi, 0,1,2,3,4,5,6,7,8,9,10,11,12,13,14,15);
}

// One wave-wide async 16B/lane copy: global (saddr + 32-bit byte offset) -> LDS.
static __device__ __forceinline__ void async_ld16(uint32_t lds_off, uint64_t base,
                                                  uint32_t byte_off) {
  asm volatile("global_load_async_to_lds_b128 %0, %1, %2"
               :: "v"(lds_off), "v"(byte_off), "s"(base) : "memory");
}

// ---------------------------------------------------------------------------
// Projection: q/k/v = pw @ (dw * x), fp32 math, f16 stores.
// Qt[b][n][c], Kt[b][n][c] (transposed), V[b][c][n] (direct).
// ---------------------------------------------------------------------------
__global__ __launch_bounds__(256) void pam_proj(
    const float* __restrict__ x,
    const float* __restrict__ dwq, const float* __restrict__ pwq,
    const float* __restrict__ dwk, const float* __restrict__ pwk,
    const float* __restrict__ dwv, const float* __restrict__ pwv,
    _Float16* __restrict__ Qt, _Float16* __restrict__ Kt,
    _Float16* __restrict__ Vv) {
  __shared__ float xs[kC][kTM];  // 32 KB
  const int tid = threadIdx.x;
  const int b = blockIdx.y;
  const int n0 = blockIdx.x * kTM;

  for (int i = tid; i < kC * kTM; i += 256) {
    int c = i >> 6, n = i & 63;
    xs[c][n] = x[((size_t)b * kC + c) * kN + n0 + n];
  }
  __syncthreads();

  const int co = tid & 127;
  const int nh = tid >> 7;
  const float* dws[3] = {dwq, dwk, dwv};
  const float* pws[3] = {pwq, pwk, pwv};

  for (int p = 0; p < 3; ++p) {
    float acc[32];
#pragma unroll
    for (int j = 0; j < 32; ++j) acc[j] = 0.f;
    const float* pw = pws[p] + (size_t)co * kC;
    const float* dw = dws[p];
    for (int c = 0; c < kC; ++c) {
      float w = pw[c] * dw[c];
#pragma unroll
      for (int j = 0; j < 32; ++j) acc[j] += w * xs[c][nh * 32 + j];
    }
    if (p == 2) {  // V: [b][co][n]
      _Float16* dst = Vv + ((size_t)b * kC + co) * kN + n0 + nh * 32;
#pragma unroll
      for (int j = 0; j < 32; ++j) dst[j] = (_Float16)acc[j];
    } else {       // Qt/Kt: [b][n][co]
      _Float16* base = (p == 0 ? Qt : Kt) +
                       ((size_t)b * kN + n0 + nh * 32) * kC + co;
#pragma unroll
      for (int j = 0; j < 32; ++j) base[(size_t)j * kC] = (_Float16)acc[j];
    }
  }
}

// ---------------------------------------------------------------------------
// Flash attention, double-buffered async tile streaming.
// ---------------------------------------------------------------------------
__global__ __launch_bounds__(256) void pam_flash(
    const _Float16* __restrict__ Qt, const _Float16* __restrict__ Kt,
    const _Float16* __restrict__ Vg, const float* __restrict__ x,
    const float* __restrict__ gamma, float* __restrict__ out) {
  __shared__ __attribute__((aligned(32))) _Float16 Kts[2][kTM][kC];   // 2x16 KB, [m][c]
  __shared__ __attribute__((aligned(32))) _Float16 Vs[2][kC][kTM];    // 2x16 KB, [c][m]
  __shared__ __attribute__((aligned(32))) _Float16 Ps[8][16][72];     // per-wave P

  const int tid = threadIdx.x;
  const int wid = tid >> 5;
  const int lane = tid & 31;
  const int l16 = lane & 15;
  const int h = lane >> 4;
  const int b = blockIdx.y;
  const int n0 = blockIdx.x * kBN + wid * 16;

  const uint64_t KtB = (uint64_t)(uintptr_t)(Kt + (size_t)b * kN * kC);
  const uint64_t VB  = (uint64_t)(uintptr_t)(Vg + (size_t)b * kC * kN);
  const uint32_t ldsK[2] = {(uint32_t)(uintptr_t)&Kts[0][0][0],
                            (uint32_t)(uintptr_t)&Kts[1][0][0]};
  const uint32_t ldsV[2] = {(uint32_t)(uintptr_t)&Vs[0][0][0],
                            (uint32_t)(uintptr_t)&Vs[1][0][0]};

  // Persistent Q fragments for this wave's 16 rows (K-dim = C = 4 chunks of 32).
  const _Float16* qrow = Qt + ((size_t)b * kN + n0 + l16) * kC;
  v16h qf[4];
#pragma unroll
  for (int kc = 0; kc < 4; ++kc) qf[kc] = frag_a(qrow, 32 * kc, h);

  v8f o[8];
#pragma unroll
  for (int i = 0; i < 8; ++i) o[i] = {};
  float rm[8], rl[8];
#pragma unroll
  for (int j = 0; j < 8; ++j) { rm[j] = -1e30f; rl[j] = 0.f; }

  // Issue one (K,V) tile: 8 async b128 instructions per wave.
  auto issue_tile = [&](int m0, int buf) {
    const uint32_t koff = (uint32_t)((size_t)m0 * kC * 2);  // contiguous 16 KB
#pragma unroll
    for (int i = 0; i < 4; ++i) {
      uint32_t idx = (uint32_t)tid + 256u * i;  // 0..1023 sixteen-byte chunks
      async_ld16(ldsK[buf] + idx * 16, KtB, koff + idx * 16);
      uint32_t row = idx >> 3, col = idx & 7;   // V tile: 128 rows x 64 halves
      async_ld16(ldsV[buf] + idx * 16, VB,
                 (row * (uint32_t)kN + (uint32_t)m0 + col * 8) * 2);
    }
  };

  constexpr int NT = kN / kTM;  // 64 tiles
  issue_tile(0, 0);

  for (int t = 0; t < NT; ++t) {
    const int buf = t & 1;
    if (t + 1 < NT) {
      issue_tile((t + 1) * kTM, buf ^ 1);  // safe: barrier at end of t-1
      asm volatile("s_wait_asynccnt 8" ::: "memory");  // tile t's 8 done (in-order)
    } else {
      asm volatile("s_wait_asynccnt 0" ::: "memory");
    }
    __syncthreads();  // all waves' shares of tile t visible

    // S[16 x 64] = Q^T K : 4 col-subtiles x 4 K-chunks.
    v8f s[4];
#pragma unroll
    for (int mt = 0; mt < 4; ++mt) {
      s[mt] = {};
      const _Float16* brow = &Kts[buf][mt * 16 + l16][0];
#pragma unroll
      for (int kc = 0; kc < 4; ++kc) {
        v16h bf = frag_b(brow, 32 * kc, h);
        s[mt] = __builtin_amdgcn_wmma_f32_16x16x32_f16(
            false, qf[kc], false, bf, (short)0, s[mt], false, false);
      }
    }

    // Online softmax. D-layout: lane l, vgpr j -> row j+8*h, col (l&15).
#pragma unroll
    for (int j = 0; j < 8; ++j) {
      float v = fmaxf(fmaxf(s[0][j], s[1][j]), fmaxf(s[2][j], s[3][j]));
      v = fmaxf(v, __shfl_xor(v, 1, 32));
      v = fmaxf(v, __shfl_xor(v, 2, 32));
      v = fmaxf(v, __shfl_xor(v, 4, 32));
      v = fmaxf(v, __shfl_xor(v, 8, 32));
      float nm = fmaxf(rm[j], v);
      float alpha = __expf(rm[j] - nm);
      rm[j] = nm;
      float ls = 0.f;
#pragma unroll
      for (int mt = 0; mt < 4; ++mt) {
        float p = __expf(s[mt][j] - nm);
        ls += p;
        Ps[wid][j + 8 * h][mt * 16 + l16] = (_Float16)p;
      }
      ls += __shfl_xor(ls, 1, 32);
      ls += __shfl_xor(ls, 2, 32);
      ls += __shfl_xor(ls, 4, 32);
      ls += __shfl_xor(ls, 8, 32);
      rl[j] = alpha * rl[j] + ls;
#pragma unroll
      for (int ct = 0; ct < 8; ++ct) o[ct][j] *= alpha;
    }

    // Intra-wave cross-lane store->load hazard on Ps.
    asm volatile("s_wait_dscnt 0" ::: "memory");

    // O[16 x 128] += P[16 x 64] * V^T[64 x 128] : 8 c-subtiles x 2 K-chunks.
    const _Float16* parow = &Ps[wid][l16][0];
    v16h pf0 = frag_a(parow, 0, h);
    v16h pf1 = frag_a(parow, 32, h);
#pragma unroll
    for (int ct = 0; ct < 8; ++ct) {
      const _Float16* brow = &Vs[buf][ct * 16 + l16][0];
      v16h b0 = frag_b(brow, 0, h);
      v16h b1 = frag_b(brow, 32, h);
      o[ct] = __builtin_amdgcn_wmma_f32_16x16x32_f16(
          false, pf0, false, b0, (short)0, o[ct], false, false);
      o[ct] = __builtin_amdgcn_wmma_f32_16x16x32_f16(
          false, pf1, false, b1, (short)0, o[ct], false, false);
    }
    __syncthreads();  // everyone done reading buf before it is refilled
  }

  // Epilogue: out[b][c][n] = gamma * O/l + x
  const float g = gamma[0];
#pragma unroll
  for (int ct = 0; ct < 8; ++ct) {
    int c = ct * 16 + l16;
#pragma unroll
    for (int j = 0; j < 8; ++j) {
      int n = n0 + j + 8 * h;
      size_t idx = ((size_t)b * kC + c) * (size_t)kN + n;
      out[idx] = g * (o[ct][j] / rl[j]) + x[idx];
    }
  }
}

extern "C" void kernel_launch(void* const* d_in, const int* in_sizes, int n_in,
                              void* d_out, int out_size, void* d_ws, size_t ws_size,
                              hipStream_t stream) {
  const float* x     = (const float*)d_in[0];
  const float* dwq   = (const float*)d_in[1];
  const float* pwq   = (const float*)d_in[2];
  const float* dwk   = (const float*)d_in[3];
  const float* pwk   = (const float*)d_in[4];
  const float* dwv   = (const float*)d_in[5];
  const float* pwv   = (const float*)d_in[6];
  const float* gamma = (const float*)d_in[7];
  float* out = (float*)d_out;

  // Workspace: Qt, Kt [B][N][C] f16 ; V [B][C][N] f16  (8 MB each, 24 MB total)
  _Float16* Qt = (_Float16*)d_ws;
  _Float16* Kt = Qt + (size_t)kB * kN * kC;
  _Float16* Vv = Kt + (size_t)kB * kN * kC;

  pam_proj<<<dim3(kN / kTM, kB), 256, 0, stream>>>(
      x, dwq, pwq, dwk, pwk, dwv, pwv, Qt, Kt, Vv);
  pam_flash<<<dim3(kN / kBN, kB), 256, 0, stream>>>(Qt, Kt, Vv, x, gamma, out);
}
